// MultiheadRCDA_23622320128192
// MI455X (gfx1250) — compile-verified, matrix operand order checked by reading, not run on hardware
//
#include <hip/hip_runtime.h>

// ---------------------------------------------------------------------------
// MultiheadRCDA for MI455X (gfx1250, wave32, WMMA).
//
// Shapes: N=8, L=300, E=256, G=8 heads, hd=32, H=W=128.
//
// Design (see analysis): pool-before-project for keys; bf16 WMMA
// (v_wmma_f32_16x16x32_bf16, fp32 accum) for the two big GEMMs (v projection
// 17 GF, col-attend 20 GF) so compute stays under the ~20us HBM roofline
// (410 MB @ 23.3 TB/s); fp32 WMMA (16x16x4) for softmax-feeding and final
// projections. 64 MB bf16 v tensor is L2-resident (192 MB L2) across its 19x
// reuse in the contraction kernel.
// ---------------------------------------------------------------------------

#define N_   8
#define L_   300
#define E_   256
#define G_   8
#define HD_  32
#define HW_  128           // H == W == 128
#define LTILES 19          // ceil(300/16)

typedef __attribute__((ext_vector_type(2)))  float  v2f;
typedef __attribute__((ext_vector_type(8)))  float  v8f;
typedef __attribute__((ext_vector_type(16))) __bf16 v16bf;

static __device__ __forceinline__ int imin(int a, int b) { return a < b ? a : b; }

// D = A(16x4) * B(4x16) + C, fp32.  A/B: 2 floats/lane (lanes0-15: K={0,1},
// lanes16-31: K={2,3}); C/D: v8f, row M = r + 8*(lane>=16), col N = lane&15.
static __device__ __forceinline__ v8f wmma_f32x4(v2f a, v2f b, v8f c) {
  return __builtin_amdgcn_wmma_f32_16x16x4_f32(false, a, false, b, (short)0, c,
                                               false, false);
}

// D = A(16x32 bf16) * B(32x16 bf16) + C(f32).  Per-lane 16 bf16 as two runs
// of 8: lanes0-15 K={0..7,16..23}, lanes16-31 K={8..15,24..31} (ISA 7.12.2).
static __device__ __forceinline__ v8f wmma_bf16(v16bf a, v16bf b, v8f c) {
  return __builtin_amdgcn_wmma_f32_16x16x32_bf16(false, a, false, b, (short)0,
                                                 c, false, false);
}

// ---------------------------------------------------------------------------
// Kernel 1a: kr_pool[n,w,e] = mean_h key_row[n,h,w,e]   (134 MB stream)
// ---------------------------------------------------------------------------
__global__ void pool_over_h(const float* __restrict__ in, float* __restrict__ out) {
  const int idx = blockIdx.x * blockDim.x + threadIdx.x;   // n*W*E + w*E + e
  const int e = idx & (E_ - 1);
  const int w = (idx >> 8) & (HW_ - 1);
  const int n = idx >> 15;
  const float* p = in + ((size_t)n * HW_ * HW_ + w) * E_ + e;
  float s = 0.f;
#pragma unroll 4
  for (int h = 0; h < HW_; ++h) s += p[(size_t)h * HW_ * E_];
  out[idx] = s * (1.0f / HW_);
}

// Kernel 1b: kc_pool[n,h,e] = mean_w key_col[n,h,w,e]
__global__ void pool_over_w(const float* __restrict__ in, float* __restrict__ out) {
  const int idx = blockIdx.x * blockDim.x + threadIdx.x;   // n*H*E + h*E + e
  const int e = idx & (E_ - 1);
  const int h = (idx >> 8) & (HW_ - 1);
  const int n = idx >> 15;
  const float* p = in + ((size_t)(n * HW_ + h) * HW_) * E_ + e;
  float s = 0.f;
#pragma unroll 4
  for (int w = 0; w < HW_; ++w) s += p[(size_t)w * E_];
  out[idx] = s * (1.0f / HW_);
}

// ---------------------------------------------------------------------------
// Kernel 2: fp32->bf16 convert (weights for v projection)
// ---------------------------------------------------------------------------
__global__ void cvt_bf16(const float* __restrict__ in, __bf16* __restrict__ out) {
  const int i = blockIdx.x * blockDim.x + threadIdx.x;
  out[i] = (__bf16)in[i];
}

// ---------------------------------------------------------------------------
// Kernel 3: generic fp32 GEMM, out[m, c] = (sum_k X[m,k]*Wt[c,k] + bias[c])*scale
// K = 256 fixed. Block = 256 thr (8 waves); grid.x = M/16; wave j covers
// column tiles j and j+8.  Uses v_wmma_f32_16x16x4_f32.
// ---------------------------------------------------------------------------
__global__ void gemm256_f32(const float* __restrict__ X, const float* __restrict__ Wt,
                            const float* __restrict__ bias, float scale,
                            float* __restrict__ out) {
  const int lane = threadIdx.x & 31;
  const int wave = threadIdx.x >> 5;
  const int hl   = lane >> 4;
  const int row0 = blockIdx.x * 16;
  const int m    = row0 + (lane & 15);
  const int koff = hl * 2;
#pragma unroll
  for (int it = 0; it < 2; ++it) {
    const int ct  = wave + it * 8;
    const int col = ct * 16 + (lane & 15);
    v8f c = {};
#pragma unroll 8
    for (int k0 = 0; k0 < E_; k0 += 4) {
      v2f a, b;
      a.x = X [(size_t)m   * E_ + k0 + koff];
      a.y = X [(size_t)m   * E_ + k0 + koff + 1];
      b.x = Wt[(size_t)col * E_ + k0 + koff];
      b.y = Wt[(size_t)col * E_ + k0 + koff + 1];
      c = wmma_f32x4(a, b, c);
    }
    const float bc = bias[col];
#pragma unroll
    for (int r = 0; r < 8; ++r) {
      const int row = row0 + r + hl * 8;
      out[(size_t)row * E_ + col] = (c[r] + bc) * scale;
    }
  }
}

// ---------------------------------------------------------------------------
// Kernel 4: v projection, bf16 WMMA. Rows M = (n,h,w) flattened (131072),
// K = 256. A row-tile staged in LDS as bf16 (shared by all 8 waves -> value
// read once from HBM). Output bf16 in natural (n,h,w,e) layout.
// ---------------------------------------------------------------------------
__global__ void vproj_bf16(const float* __restrict__ V, const __bf16* __restrict__ Wb,
                           const float* __restrict__ bias, __bf16* __restrict__ out) {
  __shared__ __bf16 lA[16 * 264];            // 16 rows x 256 K, padded stride
  const int t    = threadIdx.x;
  const int row0 = blockIdx.x * 16;
  {
    const int row = t >> 4, c0 = (t & 15) * 16;
    const float* src = V + (size_t)(row0 + row) * E_ + c0;
#pragma unroll
    for (int i = 0; i < 16; ++i) lA[row * 264 + c0 + i] = (__bf16)src[i];
  }
  __syncthreads();
  const int lane = t & 31, wave = t >> 5, hl = lane >> 4;
#pragma unroll
  for (int it = 0; it < 2; ++it) {
    const int ct  = wave + it * 8;
    const int col = ct * 16 + (lane & 15);
    v8f c = {};
#pragma unroll
    for (int kf = 0; kf < 8; ++kf) {
      const int kb = kf * 32 + hl * 8;
      v16bf a, b;
      const __bf16* pa = &lA[(lane & 15) * 264 + kb];
      const __bf16* pb = Wb + (size_t)col * E_ + kb;
#pragma unroll
      for (int i = 0; i < 8; ++i) {
        a[i] = pa[i];  a[8 + i] = pa[16 + i];
        b[i] = pb[i];  b[8 + i] = pb[16 + i];
      }
      c = wmma_bf16(a, b, c);
    }
    const float bc = bias[col];
#pragma unroll
    for (int r = 0; r < 8; ++r)
      out[(size_t)(row0 + r + hl * 8) * E_ + col] = (__bf16)(c[r] + bc);
  }
}

// ---------------------------------------------------------------------------
// Kernel 5: logits (fp32 WMMA, K=32) + row softmax over 128 keys.
// Grid (ltile, g, n), block 256 (wave j -> key tile j). Writes either fp32
// (attn_row) or bf16 (attn_col, consumed as WMMA A operand).
// ---------------------------------------------------------------------------
__global__ void attn_scores_softmax(const float* __restrict__ Q, const float* __restrict__ K,
                                    float* __restrict__ outF, __bf16* __restrict__ outB) {
  __shared__ float lg[16 * 129];
  __shared__ float red[16 * 16];
  const int lt = blockIdx.x, g = blockIdx.y, n = blockIdx.z;
  const int l0 = lt * 16;
  const int t = threadIdx.x, lane = t & 31, wave = t >> 5, hl = lane >> 4;
  {
    const int ml = imin(l0 + (lane & 15), L_ - 1);     // clamp partial tile
    const int koff = hl * 2;
    const int w  = wave * 16 + (lane & 15);
    const float* qp = Q + ((size_t)n * L_  + ml) * E_ + g * HD_;
    const float* kp = K + ((size_t)n * HW_ + w)  * E_ + g * HD_;
    v8f c = {};
#pragma unroll
    for (int k0 = 0; k0 < HD_; k0 += 4) {
      v2f a, b;
      a.x = qp[k0 + koff]; a.y = qp[k0 + koff + 1];
      b.x = kp[k0 + koff]; b.y = kp[k0 + koff + 1];
      c = wmma_f32x4(a, b, c);
    }
#pragma unroll
    for (int r = 0; r < 8; ++r)
      lg[(r + hl * 8) * 129 + wave * 16 + (lane & 15)] = c[r];
  }
  __syncthreads();
  // softmax: thread t owns (row = t>>4, cols seg*8..seg*8+7)
  const int row = t >> 4, seg = t & 15;
  float loc[8], mx = -3.4e38f;
#pragma unroll
  for (int i = 0; i < 8; ++i) { loc[i] = lg[row * 129 + seg * 8 + i]; mx = fmaxf(mx, loc[i]); }
  red[row * 16 + seg] = mx;
  __syncthreads();
  float rmx = -3.4e38f;
#pragma unroll
  for (int i = 0; i < 16; ++i) rmx = fmaxf(rmx, red[row * 16 + i]);
  float s = 0.f;
#pragma unroll
  for (int i = 0; i < 8; ++i) { loc[i] = __expf(loc[i] - rmx); s += loc[i]; }
  __syncthreads();
  red[row * 16 + seg] = s;
  __syncthreads();
  float tot = 0.f;
#pragma unroll
  for (int i = 0; i < 16; ++i) tot += red[row * 16 + i];
  const float inv = 1.0f / tot;
  const int l = l0 + row;
  if (l < L_) {
    const size_t base = (((size_t)n * G_ + g) * L_ + l) * HW_ + seg * 8;
    if (outF) {
#pragma unroll
      for (int i = 0; i < 8; ++i) outF[base + i] = loc[i] * inv;
    } else {
#pragma unroll
      for (int i = 0; i < 8; ++i) outB[base + i] = (__bf16)(loc[i] * inv);
    }
  }
}

// ---------------------------------------------------------------------------
// Kernel 6: fused col-attend + row-weighting (the 20 GFLOP contraction).
//   T[l,w,d]  = sum_h acol[l,h] * v[h,w,g*32+d]     (bf16 WMMA, K=128)
//   out[l,d] += arow[l,w] * T[l,w,d]                (VALU fold)
// Block = 128 thr (4 waves = 4 L-tiles sharing the per-w v tile through LDS,
// stored transposed so bf16 B-fragments are two contiguous 16B LDS reads).
// Grid (ltile-group of 4, g, n). Output tmp layout (l, n, e=g*32+d).
// ---------------------------------------------------------------------------
__global__ void rc_attend(const __bf16* __restrict__ Acol, const float* __restrict__ Arow,
                          const __bf16* __restrict__ Vb, float* __restrict__ outT) {
  __shared__ __bf16 vt[32 * 132];            // [d][h] transposed, padded
  const int g = blockIdx.y, n = blockIdx.z;
  const int ng = n * G_ + g;
  const int t = threadIdx.x, lane = t & 31, wave = t >> 5, hl = lane >> 4;
  const int ltile = blockIdx.x * 4 + wave;
  const bool live = (ltile < LTILES);
  const int l0 = imin(ltile, LTILES - 1) * 16;

  // Preload attn_col A-fragments (16 l-rows x 128 h) - invariant over w.
  v16bf afr[4];
  {
    const int m = imin(l0 + (lane & 15), L_ - 1);
    const __bf16* ap = Acol + ((size_t)ng * L_ + m) * HW_;
#pragma unroll
    for (int kf = 0; kf < 4; ++kf) {
      const int kb = kf * 32 + hl * 8;
#pragma unroll
      for (int i = 0; i < 8; ++i) {
        afr[kf][i]     = ap[kb + i];
        afr[kf][8 + i] = ap[kb + 16 + i];
      }
    }
  }
  float acc1[8] = {0.f}, acc2[8] = {0.f};
  const float* arp = Arow + (size_t)ng * L_ * HW_;

  for (int w = 0; w < HW_; ++w) {
    { // cooperative load of v[:,w,g-slice] (128h x 32d), transposed to [d][h]
      const int h = t;
      const __bf16* src = Vb + (((size_t)n * HW_ + h) * HW_ + w) * E_ + g * HD_;
#pragma unroll
      for (int d = 0; d < 32; ++d) vt[d * 132 + h] = src[d];
    }
    __syncthreads();
    v8f c1 = {}, c2 = {};
#pragma unroll
    for (int kf = 0; kf < 4; ++kf) {
      const int kb = kf * 32 + hl * 8;
      v16bf b1, b2;
      const __bf16* p1 = &vt[(lane & 15) * 132 + kb];         // cols d=0..15
      const __bf16* p2 = &vt[(16 + (lane & 15)) * 132 + kb];  // cols d=16..31
#pragma unroll
      for (int i = 0; i < 8; ++i) {
        b1[i] = p1[i];  b1[8 + i] = p1[16 + i];
        b2[i] = p2[i];  b2[8 + i] = p2[16 + i];
      }
      c1 = wmma_bf16(afr[kf], b1, c1);
      c2 = wmma_bf16(afr[kf], b2, c2);
    }
#pragma unroll
    for (int r = 0; r < 8; ++r) {
      const int l = imin(l0 + r + hl * 8, L_ - 1);
      const float a = arp[(size_t)l * HW_ + w];
      acc1[r] += a * c1[r];
      acc2[r] += a * c2[r];
    }
    __syncthreads();
  }
  if (live) {
#pragma unroll
    for (int r = 0; r < 8; ++r) {
      const int l = l0 + r + hl * 8;
      if (l < L_) {
        const size_t base = ((size_t)l * N_ + n) * E_ + g * HD_;
        outT[base + (lane & 15)]      = acc1[r];
        outT[base + 16 + (lane & 15)] = acc2[r];
      }
    }
  }
}

// ---------------------------------------------------------------------------
// Launcher
// ---------------------------------------------------------------------------
extern "C" void kernel_launch(void* const* d_in, const int* in_sizes, int n_in,
                              void* d_out, int out_size, void* d_ws, size_t ws_size,
                              hipStream_t stream) {
  (void)in_sizes; (void)n_in; (void)out_size; (void)ws_size;
  const float* query_row = (const float*)d_in[0];
  const float* query_col = (const float*)d_in[1];
  const float* key_row   = (const float*)d_in[2];
  const float* key_col   = (const float*)d_in[3];
  const float* value     = (const float*)d_in[4];
  const float* ipw       = (const float*)d_in[5];   // (1280, 256)
  const float* ipb       = (const float*)d_in[6];   // (1280,)
  const float* opw       = (const float*)d_in[7];   // (256, 256)
  const float* opb       = (const float*)d_in[8];   // (256,)
  // d_in[9] = num_heads (constant 8, hardcoded)

  // workspace carve (~94 MB total)
  char* ws = (char*)d_ws;
  size_t off = 0;
  auto alloc = [&](size_t bytes) -> void* {
    void* p = ws + off;
    off += (bytes + 255) & ~(size_t)255;
    return p;
  };
  float*  kr_pool = (float*) alloc((size_t)N_ * HW_ * E_ * 4);
  float*  kc_pool = (float*) alloc((size_t)N_ * HW_ * E_ * 4);
  float*  qr      = (float*) alloc((size_t)N_ * L_  * E_ * 4);
  float*  qc      = (float*) alloc((size_t)N_ * L_  * E_ * 4);
  float*  krp     = (float*) alloc((size_t)N_ * HW_ * E_ * 4);
  float*  kcp     = (float*) alloc((size_t)N_ * HW_ * E_ * 4);
  float*  arow    = (float*) alloc((size_t)N_ * G_ * L_ * HW_ * 4);
  __bf16* acol    = (__bf16*)alloc((size_t)N_ * G_ * L_ * HW_ * 2);
  __bf16* w4b     = (__bf16*)alloc((size_t)E_ * E_ * 2);
  __bf16* vbf     = (__bf16*)alloc((size_t)N_ * HW_ * HW_ * E_ * 2);  // 64 MB, L2-resident
  float*  tmp     = (float*) alloc((size_t)L_ * N_ * E_ * 4);

  const float qscale = 0.17677669529663687f;   // hd^-0.5 = 32^-0.5

  // 1) pool keys before projection (linearity)
  pool_over_h<<<(N_ * HW_ * E_) / 256, 256, 0, stream>>>(key_row, kr_pool);
  pool_over_w<<<(N_ * HW_ * E_) / 256, 256, 0, stream>>>(key_col, kc_pool);
  // 2) bf16 weights for v projection
  cvt_bf16<<<(E_ * E_) / 256, 256, 0, stream>>>(ipw + 4 * E_ * E_, w4b);
  // 3) small fp32-WMMA projections
  gemm256_f32<<<(N_ * L_)  / 16, 256, 0, stream>>>(query_row, ipw,               ipb,          qscale, qr);
  gemm256_f32<<<(N_ * L_)  / 16, 256, 0, stream>>>(query_col, ipw + 1 * E_ * E_, ipb + 1 * E_, qscale, qc);
  gemm256_f32<<<(N_ * HW_) / 16, 256, 0, stream>>>(kr_pool,   ipw + 2 * E_ * E_, ipb + 2 * E_, 1.0f,   krp);
  gemm256_f32<<<(N_ * HW_) / 16, 256, 0, stream>>>(kc_pool,   ipw + 3 * E_ * E_, ipb + 3 * E_, 1.0f,   kcp);
  // 4) big bf16 v projection (17 GFLOP)
  vproj_bf16<<<(N_ * HW_ * HW_) / 16, 256, 0, stream>>>(value, w4b, ipb + 4 * E_, vbf);
  // 5) logits + softmax
  attn_scores_softmax<<<dim3(LTILES, G_, N_), 256, 0, stream>>>(qr, krp, arow, nullptr);
  attn_scores_softmax<<<dim3(LTILES, G_, N_), 256, 0, stream>>>(qc, kcp, nullptr, acol);
  // 6) fused col-attend + row-weighting (20 GFLOP bf16 WMMA)
  rc_attend<<<dim3((LTILES + 3) / 4, G_, N_), 128, 0, stream>>>(acol, arow, vbf, tmp);
  // 7) output projection -> d_out (L, N, E)
  gemm256_f32<<<(L_ * N_) / 16, 256, 0, stream>>>(tmp, opw, opb, 1.0f, (float*)d_out);
}
